// HLFRN_29326036697159
// MI455X (gfx1250) — matrix operand order repaired and checked by reading, not compiled, not run on hardware
//
#include <hip/hip_runtime.h>
#include <math.h>

// ---------------------------------------------------------------------------
// MI455X (gfx1250, wave32). All GEMM-shaped work on v_wmma_f32_16x16x32_bf16.
// Fragment layouts used here (wave32; contiguous k, not the ISA table order), exploited for vector loads:
//   A 16x32 bf16 : m = lane&15 ; elems 0..15 = row[hs*16 .. hs*16+15]
//                  -> two aligned 16B loads from a K-contiguous row.
//   B 32x16 bf16 : n = lane&15 ; elems 0..15 = col_n[hs*16 .. hs*16+15]
//                  -> two aligned 16B loads from a K-contiguous column.
//   C/D 16x16 f32: elem r -> M = r + 8*hs, N = lane&15.
// ---------------------------------------------------------------------------

#define ANGN   9
#define C_IN   3
#define HD     128
#define NF     16
#define UV     (ANGN*ANGN)      // 81
#define HW     (HD*HD)          // 16384
#define CHW    (C_IN*HW)        // 49152
#define NSLICE (UV*C_IN)        // 243
#define PI_F   3.14159265358979323846f

typedef __attribute__((ext_vector_type(16))) __bf16 v16bf;
typedef __attribute__((ext_vector_type(8)))  __bf16 v8bf;
typedef __attribute__((ext_vector_type(8)))  float  v8f;

__device__ __forceinline__ v16bf load_b16(const __bf16* p) {
  v8bf lo = *(const v8bf*)p;
  v8bf hi = *(const v8bf*)(p + 8);
  return __builtin_shufflevector(lo, hi, 0,1,2,3,4,5,6,7,8,9,10,11,12,13,14,15);
}
__device__ __forceinline__ v16bf load_a16(const __bf16* rowbase, int hs) {
  v8bf lo = *(const v8bf*)(rowbase + hs * 16);
  v8bf hi = *(const v8bf*)(rowbase + hs * 16 + 8);
  return __builtin_shufflevector(lo, hi, 0,1,2,3,4,5,6,7,8,9,10,11,12,13,14,15);
}
__device__ __forceinline__ v8f wmma_bf16(v16bf a, v16bf b, v8f c) {
  return __builtin_amdgcn_wmma_f32_16x16x32_bf16(false, a, false, b,
                                                 (short)0, c, false, false);
}
__device__ __forceinline__ void band_pos(int band, int l, int* u, int* v) {
  if (band <= 8) { *u = l; *v = band - l; }
  else           { int j = band - 8; *u = j + l; *v = 8 - l; }
}

// ---------------------------------------------------------------------------
// DCT matrices: Ds (128x128 bf16, + transpose), Da (9x9 f32),
// Kron(Da,Da) zero-padded to 96x96 bf16 (+ transpose).
// ---------------------------------------------------------------------------
__global__ void k_build_dct(__bf16* dsB, __bf16* dsTB, float* daF,
                            __bf16* kron, __bf16* kronT) {
  int t = threadIdx.x;
  const float is2 = 0.70710678118654752440f;
  for (int i = t; i < HD * HD; i += blockDim.x) {
    int k = i >> 7, n = i & 127;
    float v = __cosf(PI_F * (n + 0.5f) * k / (float)HD) * sqrtf(2.0f / HD);
    if (k == 0) v *= is2;
    dsB[i] = (__bf16)v;
    dsTB[n * HD + k] = (__bf16)v;
  }
  for (int i = t; i < ANGN * ANGN; i += blockDim.x) {
    int k = i / ANGN, n = i % ANGN;
    float v = __cosf(PI_F * (n + 0.5f) * k / (float)ANGN) * sqrtf(2.0f / ANGN);
    if (k == 0) v *= is2;
    daF[i] = v;
  }
  __syncthreads();
  for (int i = t; i < 96 * 96; i += blockDim.x) {
    int r = i / 96, c = i % 96;
    float f = 0.0f, g = 0.0f;
    if (r < UV && c < UV) {
      int U = r / ANGN, V = r % ANGN, u = c / ANGN, v = c % ANGN;
      f = daF[U * ANGN + u] * daF[V * ANGN + v];
      g = daF[u * ANGN + U] * daF[v * ANGN + V];
    }
    kron[i]  = (__bf16)f;
    kronT[i] = (__bf16)g;
  }
}

// ---------------------------------------------------------------------------
// T[slice] = M(128x128) @ X[slice]   (X f32 -> T bf16)
// X staged TRANSPOSED in LDS so B fragments are K-contiguous.
// ---------------------------------------------------------------------------
#define XP 136   // LDS pitch (bf16): 272B rows, 16B aligned, conflict-spread
__global__ void k_gemm_left(const float* __restrict__ X,
                            const __bf16* __restrict__ M,
                            __bf16* __restrict__ T) {
  __shared__ __bf16 ldsXT[HD * XP];
  int slice = blockIdx.x, t = threadIdx.x;
  const float* Xs = X + (size_t)slice * HW;
  for (int i = t; i < HW; i += blockDim.x) {
    int hh = i >> 7, ww = i & 127;
    ldsXT[ww * XP + hh] = (__bf16)Xs[i];
  }
  __syncthreads();
  int wv = t >> 5, lane = t & 31, hs = lane >> 4, nlo = lane & 15;
  __bf16* Ts = T + (size_t)slice * HW;
  for (int tt = 0; tt < 8; ++tt) {
    int tile = wv * 8 + tt, Mi = tile >> 3, Nj = tile & 7;
    v8f acc = {};
    #pragma unroll
    for (int kc = 0; kc < 4; ++kc) {
      v16bf a = load_a16(M + (Mi * 16 + nlo) * HD + kc * 32, hs);
      v16bf b = load_b16(&ldsXT[(Nj * 16 + nlo) * XP + kc * 32 + hs * 16]);
      acc = wmma_bf16(a, b, acc);
    }
    #pragma unroll
    for (int r = 0; r < 8; ++r)
      Ts[(Mi * 16 + r + 8 * hs) * HD + Nj * 16 + nlo] = (__bf16)acc[r];
  }
}

// ---------------------------------------------------------------------------
// Y[slice] = T[slice] @ G  (+ optional addSrc).  Gt[n][k] = G[k][n] is passed
// pre-transposed (it is simply the other Ds matrix) -> B = global b128 loads.
// A = T rows staged row-major in LDS (vector copy-in).
// ---------------------------------------------------------------------------
__global__ void k_gemm_right(const __bf16* __restrict__ T,
                             const __bf16* __restrict__ Gt,
                             const float* __restrict__ addSrc,
                             float* __restrict__ Y) {
  __shared__ __bf16 ldsT[HD * XP];
  int slice = blockIdx.x, t = threadIdx.x;
  const __bf16* Ts = T + (size_t)slice * HW;
  for (int i = t * 8; i < HW; i += blockDim.x * 8) {
    int row = i >> 7, col = i & 127;
    *(v8bf*)&ldsT[row * XP + col] = *(const v8bf*)(Ts + i);
  }
  __syncthreads();
  int wv = t >> 5, lane = t & 31, hs = lane >> 4, nlo = lane & 15;
  float* Ys = Y + (size_t)slice * HW;
  const float* As = addSrc ? addSrc + (size_t)slice * HW : nullptr;
  for (int tt = 0; tt < 8; ++tt) {
    int tile = wv * 8 + tt, Mi = tile >> 3, Nj = tile & 7;
    v8f acc = {};
    #pragma unroll
    for (int kc = 0; kc < 4; ++kc) {
      v16bf a = load_a16(&ldsT[(Mi * 16 + nlo) * XP + kc * 32], hs);
      v16bf b = load_b16(Gt + (Nj * 16 + nlo) * HD + kc * 32 + hs * 16);
      acc = wmma_bf16(a, b, acc);
    }
    #pragma unroll
    for (int r = 0; r < 8; ++r) {
      int row = Mi * 16 + r + 8 * hs, col = Nj * 16 + nlo;
      float v = acc[r];
      if (As) v += As[row * HD + col];
      Ys[row * HD + col] = v;
    }
  }
}

// ---------------------------------------------------------------------------
// Y[0:81][N] = A(96x96 zero-padded) @ X[0:81][N], N = CHW.
// Block stages a transposed, ZERO-PADDED [128 cols][96 k] B panel in LDS
// (pad must be zero: 0 x NaN = NaN under WMMA). 8 waves x 6 M-tiles each.
// ---------------------------------------------------------------------------
#define AP 104   // 208B pitch, 16B aligned
__global__ void k_ang_gemm(const float* __restrict__ X,
                           const __bf16* __restrict__ A,
                           float* __restrict__ Y) {
  __shared__ __bf16 ldsBT[HD * AP];
  int nb = blockIdx.x * 128, t = threadIdx.x;
  for (int i = t; i < 96 * 128; i += blockDim.x) {
    int k = i >> 7, c = i & 127;
    float v = (k < UV) ? X[(size_t)k * CHW + nb + c] : 0.0f;
    ldsBT[c * AP + k] = (__bf16)v;
  }
  __syncthreads();
  int wv = t >> 5, lane = t & 31, hs = lane >> 4, nlo = lane & 15;
  int n0 = wv * 16;
  for (int mi = 0; mi < 6; ++mi) {
    v8f acc = {};
    #pragma unroll
    for (int kc = 0; kc < 3; ++kc) {
      v16bf a = load_a16(A + (mi * 16 + nlo) * 96 + kc * 32, hs);
      v16bf b = load_b16(&ldsBT[(n0 + nlo) * AP + kc * 32 + hs * 16]);
      acc = wmma_bf16(a, b, acc);
    }
    #pragma unroll
    for (int r = 0; r < 8; ++r) {
      int m = mi * 16 + r + 8 * hs;
      if (m < UV) Y[(size_t)m * CHW + nb + n0 + nlo] = acc[r];
    }
  }
}

// ---------------------------------------------------------------------------
// Unified conv3d 3x3x3 implicit GEMM.  K-index k = tap*16 + ci  (tap = 0..26,
// ci zero-padded to 16 for the init conv).  A whole B fragment has constant
// tap and ci = elem index, so LDS layout [tap/3 (+pad)][col+kw][ci] makes both
// fragments two b128 loads.  K = 432 -> 14 chunks; tap 27 row is zero-filled.
//   mode 0: dst = conv(gather xf, band) + bias                (y0)
//   mode 1: dst = src + conv(prelu(src)) + bias               (resblock)
//   mode 2: xrec scatter = src + conv(prelu(src)) + bias + y0 (final)
// ---------------------------------------------------------------------------
__global__ void k_conv(const float* __restrict__ src,   // mode0: xf[81][3][HW]; else yin[16][L][HW]
                       const float* __restrict__ w,     // mode0: [16][3][27]; else [16][16][27]
                       const float* __restrict__ bias,  // [16]
                       const float* __restrict__ aarr, int aidx,
                       int L, int band, int mode,
                       float* __restrict__ dst,
                       const float* __restrict__ y0) {
  __shared__ __bf16 ldsB[10 * 130 * 16];   // [row9 0..9][col -1..128][ci]
  __shared__ __bf16 ldsW[16 * 448];        // [co][k], zero-padded
  int blk = blockIdx.x, d = blk / HD, h = blk % HD, t = threadIdx.x;
  float alpha = (mode == 0) ? 0.0f : aarr[aidx];
  __builtin_prefetch(w, 0, 3);

  for (int i = t; i < 16 * 448; i += blockDim.x) {
    int co = i / 448, k = i % 448;
    float v = 0.0f;
    if (k < 432) {
      int tap = k >> 4, ci = k & 15;
      if (mode == 0) { if (ci < C_IN) v = w[co * 81 + ci * 27 + tap]; }
      else           v = w[co * 432 + ci * 27 + tap];
    }
    ldsW[i] = (__bf16)v;
  }
  for (int i = t; i < 10 * 130 * 16; i += blockDim.x) {
    int ci = i & 15, rest = i >> 4;
    int col = rest % 130, row9 = rest / 130;
    float v = 0.0f;
    if (row9 < 9) {
      int dz = d + row9 / 3 - 1, hz = h + row9 % 3 - 1, wz = col - 1;
      bool in = (dz >= 0 && dz < L && hz >= 0 && hz < HD && wz >= 0 && wz < HD);
      if (mode == 0) {
        if (ci < C_IN && in) {
          int u, vv; band_pos(band, dz, &u, &vv);
          v = src[((size_t)(u * ANGN + vv) * C_IN + ci) * HW + hz * HD + wz];
        }
      } else if (in) {
        v = src[((size_t)ci * L + dz) * HW + hz * HD + wz];
        v = (v >= 0.0f) ? v : alpha * v;       // PReLU fused into staging
      }
    }
    ldsB[i] = (__bf16)v;
  }
  __syncthreads();

  int wv = t >> 5, lane = t & 31, hs = lane >> 4, nlo = lane & 15;
  int w0 = wv * 16;
  v8f acc = {};
  #pragma unroll
  for (int kc = 0; kc < 14; ++kc) {
    v16bf a = load_a16(&ldsW[nlo * 448 + kc * 32], hs);
    int tap = kc * 2 + hs;                     // constant over fragment elems
    int row9 = tap / 3, kw = tap - row9 * 3;
    v16bf b = load_b16(&ldsB[(row9 * 130 + (w0 + nlo + kw)) * 16]);
    acc = wmma_bf16(a, b, acc);
  }
  #pragma unroll
  for (int r = 0; r < 8; ++r) {
    int co = r + 8 * hs, wp = w0 + nlo;
    size_t idx = ((size_t)co * L + d) * HW + h * HD + wp;
    float v = acc[r] + bias[co];
    if (mode == 0)      dst[idx] = v;
    else if (mode == 1) dst[idx] = v + src[idx];
    else {
      int u, vv; band_pos(band, d, &u, &vv);
      dst[((size_t)co * UV + (u * ANGN + vv)) * HW + h * HD + wp] =
          v + src[idx] + y0[idx];
    }
  }
}

// ---------------------------------------------------------------------------
// 1x1 conv NF->3: out[uv][c][hw] = sum_n w_last[c][n] * xrec[n][uv][hw]
// ---------------------------------------------------------------------------
__global__ void k_last(const float* __restrict__ xrec,
                       const float* __restrict__ wl,
                       float* __restrict__ out) {
  size_t i = (size_t)blockIdx.x * blockDim.x + threadIdx.x;
  if (i >= (size_t)UV * HW) return;
  int uv = (int)(i / HW), hw = (int)(i % HW);
  float in[NF];
  #pragma unroll
  for (int n = 0; n < NF; ++n) in[n] = xrec[((size_t)n * UV + uv) * HW + hw];
  #pragma unroll
  for (int c = 0; c < C_IN; ++c) {
    float s = 0.0f;
    #pragma unroll
    for (int n = 0; n < NF; ++n) s += wl[c * NF + n] * in[n];
    out[((size_t)uv * C_IN + c) * HW + hw] = s;
  }
}

// ---------------------------------------------------------------------------
// Host launch
// ---------------------------------------------------------------------------
static inline int h_band_len(int b) { return (b <= 8) ? b + 1 : 17 - b; }

extern "C" void kernel_launch(void* const* d_in, const int* in_sizes, int n_in,
                              void* d_out, int out_size, void* d_ws, size_t ws_size,
                              hipStream_t stream) {
  (void)in_sizes; (void)n_in; (void)out_size; (void)ws_size;
  const float* x = (const float*)d_in[0];
  const float* w_last = (const float*)d_in[21];

  struct BP { const float *wi, *bi, *a, *wb, *bb; };
  auto gp = [&](int base) {
    return BP{ (const float*)d_in[base],     (const float*)d_in[base + 1],
               (const float*)d_in[base + 2], (const float*)d_in[base + 3],
               (const float*)d_in[base + 4] };
  };
  BP P[4] = { gp(1), gp(6), gp(11), gp(16) };   // dc, lo, mi, hi

  char* ws = (char*)d_ws;
  size_t off = 0;
  auto carve = [&](size_t bytes) {
    char* p = ws + off;
    off = (off + bytes + 255) & ~(size_t)255;
    return p;
  };
  __bf16* DsB   = (__bf16*)carve(HD * HD * 2);
  __bf16* DsTB  = (__bf16*)carve(HD * HD * 2);
  float*  DaF   = (float*) carve(ANGN * ANGN * 4);
  __bf16* Kron  = (__bf16*)carve(96 * 96 * 2);
  __bf16* KronT = (__bf16*)carve(96 * 96 * 2);
  __bf16* Tbf   = (__bf16*)carve((size_t)NSLICE * HW * 2);
  float*  xf_s  = (float*) carve((size_t)NSLICE * HW * 4);  // reused as buf1
  float*  xf    = (float*) carve((size_t)NSLICE * HW * 4);  // reused as buf2
  float*  y0b   = (float*) carve((size_t)NF * ANGN * HW * 4);
  float*  y1b   = (float*) carve((size_t)NF * ANGN * HW * 4);
  float*  xrec  = (float*) carve((size_t)NF * UV * HW * 4);

  dim3 blk256(256);

  // 0) DCT matrices
  k_build_dct<<<1, blk256, 0, stream>>>(DsB, DsTB, DaF, Kron, KronT);

  // 1) Forward spatial DCT: xf_s = Ds @ X @ Ds^T.
  //    Right-mult Gt = (Ds^T)^T = Ds row-major.
  k_gemm_left <<<NSLICE, blk256, 0, stream>>>(x, DsB, Tbf);
  k_gemm_right<<<NSLICE, blk256, 0, stream>>>(Tbf, DsB, nullptr, xf_s);

  // 2) Forward angular DCT: xf = Kron(Da,Da) @ xf_s
  k_ang_gemm<<<CHW / 128, blk256, 0, stream>>>(xf_s, Kron, xf);

  // 3) 17 frequency-band branches
  for (int band = 0; band < 17; ++band) {
    int L = h_band_len(band);
    const BP& p = (band == 0) ? P[0] : (band <= 2) ? P[1] : (band <= 8) ? P[2] : P[3];
    dim3 grid(L * HD);
    k_conv<<<grid, blk256, 0, stream>>>(xf,  p.wi, p.bi, nullptr, 0,
                                        L, band, 0, y0b, nullptr);
    k_conv<<<grid, blk256, 0, stream>>>(y0b, p.wb, p.bb, p.a, 0,
                                        L, band, 1, y1b, nullptr);
    k_conv<<<grid, blk256, 0, stream>>>(y1b, p.wb + NF * NF * 27, p.bb + NF,
                                        p.a, 1, L, band, 2, xrec, y0b);
  }

  // 4) 1x1 conv NF->3 (buf1 reuses xf_s; xf_s fully consumed in step 2)
  float* buf1 = xf_s;
  k_last<<<((size_t)UV * HW + 255) / 256, blk256, 0, stream>>>(xrec, w_last, buf1);

  // 5) Inverse angular DCT (buf2 reuses xf; xf consumed in step 3)
  float* buf2 = xf;
  k_ang_gemm<<<CHW / 128, blk256, 0, stream>>>(buf1, KronT, buf2);

  // 6) Inverse spatial DCT + residual x: out = Ds^T @ buf2 @ Ds + x.
  //    Left = Ds^T row-major; right-mult Gt = Ds^T row-major.
  k_gemm_left <<<NSLICE, blk256, 0, stream>>>(buf2, DsTB, Tbf);
  k_gemm_right<<<NSLICE, blk256, 0, stream>>>(Tbf, DsTB, x, (float*)d_out);
}
